// LinearAttention_34668976013568
// MI455X (gfx1250) — matrix operand order repaired and checked by reading, not compiled, not run on hardware
//
#include <hip/hip_runtime.h>
#include <hip/hip_fp16.h>

typedef __attribute__((ext_vector_type(16))) _Float16 v16h;
typedef __attribute__((ext_vector_type(8)))  float    v8f;

#define BATCH   16
#define CIN     256
#define HID     512        // HEAD * DIM_HEAD
#define HEADS   8
#define DHEAD   64
#define NSP     4096       // 64*64 spatial
#define GN_EPS  1e-5f

// ---------------- WMMA fragment helpers (wave32, v_wmma_f32_16x16x32_f16) ---
// A tile staged in LDS row-major [16][32] halfs (ld=32).
// ISA A-layout (16-bit, 16x32): lane m=L%16, hi=L/16;
//   elements 0..7 -> k = hi*8 + e ; elements 8..15 -> k = 16 + hi*8 + (e-8)
__device__ __forceinline__ v16h frag_a(const _Float16* sA) {
  int lane = threadIdx.x & 31;
  int m = lane & 15, hi = lane >> 4;
  const _Float16* p = sA + m * 32 + hi * 8;
  v16h a;
  ((float4*)&a)[0] = *(const float4*)(p);       // k = hi*8 + 0..7
  ((float4*)&a)[1] = *(const float4*)(p + 16);  // k = hi*8 + 16..23
  return a;
}
// B tile staged in LDS n-major [16][32] halfs (per-n k contiguous).
// ISA B-layout (16-bit, 32x16): lane n=L%16, hi=L/16;
//   elements 0..7 -> k = hi*8 + e ; elements 8..15 -> k = 16 + hi*8 + (e-8)
__device__ __forceinline__ v16h frag_b(const _Float16* sB) {
  int lane = threadIdx.x & 31;
  int n = lane & 15, hi = lane >> 4;
  const _Float16* p = sB + n * 32 + hi * 8;
  v16h b;
  ((float4*)&b)[0] = *(const float4*)(p);
  ((float4*)&b)[1] = *(const float4*)(p + 16);
  return b;
}

// ---------------- Generic 64xBN block GEMM, BK=32, 256 threads (8 waves) ----
// A:  [M][K] row-major f16 (k contiguous).
// Bt: [N][K] row-major f16 (k contiguous) -- all producers emit n-major B, so
//     staging is pure float4 copies (no transpose scatter).
// Double-buffered LDS: stage tile k+1 while WMMAs consume tile k; one barrier
// per k-step. BN=128: wave owns 2x2 16x16 tiles; BN=64: 1x2 tiles.
template <int BN, bool OUT_F32, bool OUT_TRANS>
__device__ void gemm_block(const _Float16* __restrict__ A, int lda,
                           const _Float16* __restrict__ Bt, int ldb,
                           void* __restrict__ outp, int ldo,
                           int K, int tileM, int tileN) {
  constexpr int BM  = 64;
  constexpr int TM  = (BN == 128) ? 2 : 1;   // m-tiles per wave
  constexpr int TN  = 2;                     // n-tiles per wave
  constexpr int WGN = (BN / 16) / TN;        // waves along n (2 or 4)
  constexpr int BCH = (BN * 4) / 256;        // B float4 chunks per thread (1|2)
  __shared__ _Float16 sA[2][BM * 32];
  __shared__ _Float16 sB[2][BN * 32];
  const int tid  = threadIdx.x;
  const int wave = tid >> 5;
  const int wm = wave / WGN, wn = wave % WGN;
  const int arow = tid >> 2, acol = (tid & 3) << 3;

  auto stage = [&](int buf, int k0) {
    *(float4*)&sA[buf][arow * 32 + acol] =
        *(const float4*)&A[(size_t)(tileM + arow) * lda + k0 + acol];
#pragma unroll
    for (int it = 0; it < BCH; ++it) {
      int c = tid + it * 256;
      int row = c >> 2, col = (c & 3) << 3;
      *(float4*)&sB[buf][row * 32 + col] =
          *(const float4*)&Bt[(size_t)(tileN + row) * ldb + k0 + col];
    }
  };

  v8f acc[TM][TN] = {};
  stage(0, 0);
  int pb = 0;
  for (int k0 = 0; k0 < K; k0 += 32) {
    __syncthreads();
    if (k0 + 32 < K) {
      stage(pb ^ 1, k0 + 32);
      if (k0 + 64 < K) {   // hint tile after next into cache (global_prefetch_b8)
        __builtin_prefetch(&A[(size_t)(tileM + arow) * lda + k0 + 64 + acol], 0, 3);
        __builtin_prefetch(&Bt[(size_t)(tileN + arow) * ldb + k0 + 64 + acol], 0, 3);
      }
    }
    v16h af[TM], bf[TN];
#pragma unroll
    for (int i = 0; i < TM; ++i) af[i] = frag_a(sA[pb] + (wm * TM + i) * 16 * 32);
#pragma unroll
    for (int j = 0; j < TN; ++j) bf[j] = frag_b(sB[pb] + (wn * TN + j) * 16 * 32);
#pragma unroll
    for (int i = 0; i < TM; ++i)
#pragma unroll
      for (int j = 0; j < TN; ++j)
        acc[i][j] = __builtin_amdgcn_wmma_f32_16x16x32_f16(
            false, af[i], false, bf[j], (short)0, acc[i][j], false, false);
    pb ^= 1;
  }

  // C/D layout: VGPR r -> M = (lane<16 ? r : r+8), N = lane%16
  const int lane = tid & 31;
  const int nn = lane & 15, mo = (lane >> 4) * 8;
#pragma unroll
  for (int i = 0; i < TM; ++i)
#pragma unroll
    for (int j = 0; j < TN; ++j)
#pragma unroll
      for (int r = 0; r < 8; ++r) {
        int m = tileM + (wm * TM + i) * 16 + mo + r;
        int n = tileN + (wn * TN + j) * 16 + nn;
        size_t idx = OUT_TRANS ? ((size_t)n * ldo + m) : ((size_t)m * ldo + n);
        if (OUT_F32) ((float*)outp)[idx] = acc[i][j][r];
        else         ((_Float16*)outp)[idx] = (_Float16)acc[i][j][r];
      }
}

// ---------------- GEMM wrappers -------------------------------------------
// QKV: Y[b][o][p] = sum_c Wq[o][c] * XnT[b][p][c]   (M=1536,K=256,N=4096)
__global__ void __launch_bounds__(256)
k_gemm_qkv(const _Float16* __restrict__ W, const _Float16* __restrict__ XnT,
           _Float16* __restrict__ Y) {
  int b = blockIdx.z;
  gemm_block<128, false, false>(W, CIN,
                                XnT + (size_t)b * NSP * CIN, CIN,
                                Y + (size_t)b * 3 * HID * NSP, NSP,
                                CIN, blockIdx.y * 64, blockIdx.x * 128);
}
// context[f][e] = sum_p k'[f][p] * v[e][p]; stored transposed -> ctxT[e][f]
__global__ void __launch_bounds__(256)
k_ctx(const _Float16* __restrict__ qkv, _Float16* __restrict__ ctxT) {
  int z = blockIdx.x, b = z >> 3, h = z & 7;
  const _Float16* km = qkv + (size_t)b * 3 * HID * NSP + (size_t)(HID + h * DHEAD) * NSP;
  const _Float16* vm = qkv + (size_t)b * 3 * HID * NSP + (size_t)(2 * HID + h * DHEAD) * NSP;
  gemm_block<64, false, true>(km, NSP, vm, NSP,
                              ctxT + (size_t)z * DHEAD * DHEAD, DHEAD,
                              NSP, 0, 0);
}
// attn[e][p] = sum_f ctxT[e][f] * qT[p][f]; stored transposed into
// hiddenT[b][p][h*64+e]  (M=64,K=64,N=4096)
__global__ void __launch_bounds__(256)
k_attn(const _Float16* __restrict__ qT, const _Float16* __restrict__ ctxT,
       _Float16* __restrict__ hiddenT) {
  int z = blockIdx.z, b = z >> 3, h = z & 7;
  gemm_block<128, false, true>(ctxT + (size_t)z * DHEAD * DHEAD, DHEAD,
                               qT + (size_t)z * NSP * DHEAD, DHEAD,
                               hiddenT + (size_t)b * NSP * HID + h * DHEAD, HID,
                               DHEAD, 0, blockIdx.x * 128);
}
// out-proj: Y[b][o][p] = sum_c Wo[o][c] * hiddenT[b][p][c]  (M=256,K=512) f32
__global__ void __launch_bounds__(256)
k_gemm_out(const _Float16* __restrict__ W, const _Float16* __restrict__ XT,
           float* __restrict__ Y) {
  int b = blockIdx.z;
  gemm_block<128, true, false>(W, HID,
                               XT + (size_t)b * NSP * HID, HID,
                               Y + (size_t)b * CIN * NSP, NSP,
                               HID, blockIdx.y * 64, blockIdx.x * 128);
}

// ---------------- GroupNorm 1: f32 [b][c][p] -> f16 xnT [b][p][c] ----------
// group = 8 channels * 4096 = 32768 contiguous floats; output packs the 8
// group channels of one p into a single b128 store.
__global__ void __launch_bounds__(256)
k_gn1(const float* __restrict__ x, const float* __restrict__ w,
      const float* __restrict__ bb, _Float16* __restrict__ xnT) {
  int blk = blockIdx.x, b = blk >> 5, g = blk & 31;
  int c0 = g * 8;
  size_t base = ((size_t)b * CIN + c0) * NSP;
  const int GSZ = 8 * NSP;
  int tid = threadIdx.x;
  float s = 0.f, ss = 0.f;
  for (int i = tid; i < GSZ; i += 256) { float v = x[base + i]; s += v; ss += v * v; }
  __shared__ float rs[256], rss[256];
  rs[tid] = s; rss[tid] = ss;
  __syncthreads();
  for (int o = 128; o > 0; o >>= 1) {
    if (tid < o) { rs[tid] += rs[tid + o]; rss[tid] += rss[tid + o]; }
    __syncthreads();
  }
  float mean = rs[0] / GSZ;
  float var  = rss[0] / GSZ - mean * mean;
  float rstd = rsqrtf(var + GN_EPS);
  float wc[8], bc[8];
#pragma unroll
  for (int c = 0; c < 8; ++c) { wc[c] = w[c0 + c] * rstd; bc[c] = bb[c0 + c] - mean * wc[c]; }
  for (int p = tid; p < NSP; p += 256) {
    union { float4 f4; _Float16 h[8]; } u;
#pragma unroll
    for (int c = 0; c < 8; ++c)
      u.h[c] = (_Float16)(x[base + (size_t)c * NSP + p] * wc[c] + bc[c]);
    *(float4*)&xnT[((size_t)b * NSP + p) * CIN + c0] = u.f4;
  }
}

// ---------------- GroupNorm 2: (y + out_b) -> f32 d_out --------------------
__global__ void __launch_bounds__(256)
k_gn2(const float* __restrict__ y, const float* __restrict__ w,
      const float* __restrict__ bset, const float* __restrict__ prebias,
      float* __restrict__ out) {
  int blk = blockIdx.x, b = blk >> 5, g = blk & 31;
  int c0 = g * 8;
  size_t base = ((size_t)b * CIN + c0) * NSP;
  const int GSZ = 8 * NSP;
  int tid = threadIdx.x;
  float s = 0.f, ss = 0.f;
  for (int i = tid; i < GSZ; i += 256) {
    float v = y[base + i] + prebias[c0 + (i >> 12)];
    s += v; ss += v * v;
  }
  __shared__ float rs[256], rss[256];
  rs[tid] = s; rss[tid] = ss;
  __syncthreads();
  for (int o = 128; o > 0; o >>= 1) {
    if (tid < o) { rs[tid] += rs[tid + o]; rss[tid] += rss[tid + o]; }
    __syncthreads();
  }
  float mean = rs[0] / GSZ;
  float var  = rss[0] / GSZ - mean * mean;
  float rstd = rsqrtf(var + GN_EPS);
  for (int i = tid; i < GSZ; i += 256) {
    int c = c0 + (i >> 12);
    float v = y[base + i] + prebias[c];
    out[base + i] = (v - mean) * rstd * w[c] + bset[c];
  }
}

// ---------------- Softmaxes ------------------------------------------------
// q: softmax over feature dim f (64 strided values), * DIM_HEAD^-0.5, written
// f-contiguous into qT[bh][p][f] (eight b128 stores per thread).
__global__ void __launch_bounds__(256)
k_softmax_q(const _Float16* __restrict__ qkv, _Float16* __restrict__ qT) {
  int idx = blockIdx.x * 256 + threadIdx.x;     // b,h,p flat: 16*8*4096
  int p = idx & (NSP - 1);
  int h = (idx >> 12) & 7;
  int b = idx >> 15;
  const _Float16* base = qkv + (size_t)b * 3 * HID * NSP + (size_t)h * DHEAD * NSP + p;
  float vals[DHEAD];
  float mx = -1e30f;
#pragma unroll
  for (int f = 0; f < DHEAD; ++f) {
    float v = (float)base[(size_t)f * NSP];
    vals[f] = v; mx = fmaxf(mx, v);
  }
  float sum = 0.f;
#pragma unroll
  for (int f = 0; f < DHEAD; ++f) { float e = __expf(vals[f] - mx); vals[f] = e; sum += e; }
  float inv = 0.125f / sum;                     // 1/sqrt(64) folded in
  _Float16* row = qT + ((size_t)(b * HEADS + h) * NSP + p) * DHEAD;
#pragma unroll
  for (int v8 = 0; v8 < 8; ++v8) {
    union { float4 f4; _Float16 h8[8]; } u;
#pragma unroll
    for (int e = 0; e < 8; ++e) u.h8[e] = (_Float16)(vals[v8 * 8 + e] * inv);
    ((float4*)row)[v8] = u.f4;
  }
}
// k: softmax over spatial dim (4096), one block per (b, hidden-channel) row,
// in place on the f16 qkv buffer.
__global__ void __launch_bounds__(256) k_softmax_k(_Float16* __restrict__ qkv) {
  int r = blockIdx.x, b = r >> 9, c = r & 511;
  _Float16* row = qkv + (size_t)b * 3 * HID * NSP + (size_t)(HID + c) * NSP;
  int tid = threadIdx.x;
  __shared__ float red[256];
  float mx = -1e30f;
  for (int i = tid; i < NSP; i += 256) mx = fmaxf(mx, (float)row[i]);
  red[tid] = mx; __syncthreads();
  for (int o = 128; o > 0; o >>= 1) {
    if (tid < o) red[tid] = fmaxf(red[tid], red[tid + o]);
    __syncthreads();
  }
  mx = red[0]; __syncthreads();
  float s = 0.f;
  for (int i = tid; i < NSP; i += 256) s += __expf((float)row[i] - mx);
  red[tid] = s; __syncthreads();
  for (int o = 128; o > 0; o >>= 1) {
    if (tid < o) red[tid] += red[tid + o];
    __syncthreads();
  }
  float inv = 1.f / red[0];
  for (int i = tid; i < NSP; i += 256) row[i] = (_Float16)(__expf((float)row[i] - mx) * inv);
}

// ---------------- misc -----------------------------------------------------
__global__ void k_f32_to_f16(const float* __restrict__ in, _Float16* __restrict__ out, int n) {
  int i = blockIdx.x * 256 + threadIdx.x;
  if (i < n) out[i] = (_Float16)in[i];
}

extern "C" void kernel_launch(void* const* d_in, const int* in_sizes, int n_in,
                              void* d_out, int out_size, void* d_ws, size_t ws_size,
                              hipStream_t stream) {
  const float* x     = (const float*)d_in[0];
  const float* gn1_w = (const float*)d_in[1];
  const float* gn1_b = (const float*)d_in[2];
  const float* qkv_w = (const float*)d_in[3];
  const float* out_w = (const float*)d_in[4];
  const float* out_b = (const float*)d_in[5];
  const float* gn2_w = (const float*)d_in[6];
  const float* gn2_b = (const float*)d_in[7];
  float* out = (float*)d_out;

  char* ws = (char*)d_ws;
  _Float16* wq      = (_Float16*)ws; ws += (size_t)3 * HID * CIN * 2;
  _Float16* wo      = (_Float16*)ws; ws += (size_t)CIN * HID * 2;
  _Float16* xnT     = (_Float16*)ws; ws += (size_t)BATCH * NSP * CIN * 2;
  _Float16* qkv     = (_Float16*)ws; ws += (size_t)BATCH * 3 * HID * NSP * 2;
  _Float16* qT      = (_Float16*)ws; ws += (size_t)BATCH * HID * NSP * 2;
  _Float16* ctxT    = (_Float16*)ws; ws += (size_t)BATCH * HEADS * DHEAD * DHEAD * 2;
  _Float16* hiddenT = (_Float16*)ws; ws += (size_t)BATCH * NSP * HID * 2;
  float*    ybuf    = (float*)ws;

  k_f32_to_f16<<<(3 * HID * CIN + 255) / 256, 256, 0, stream>>>(qkv_w, wq, 3 * HID * CIN);
  k_f32_to_f16<<<(CIN * HID + 255) / 256, 256, 0, stream>>>(out_w, wo, CIN * HID);

  k_gn1<<<BATCH * 32, 256, 0, stream>>>(x, gn1_w, gn1_b, xnT);

  k_gemm_qkv<<<dim3(NSP / 128, 3 * HID / 64, BATCH), 256, 0, stream>>>(wq, xnT, qkv);

  k_softmax_q<<<BATCH * HEADS * NSP / 256, 256, 0, stream>>>(qkv, qT);
  k_softmax_k<<<BATCH * HID, 256, 0, stream>>>(qkv);

  k_ctx<<<BATCH * HEADS, 256, 0, stream>>>(qkv, ctxT);
  k_attn<<<dim3(NSP / 128, 1, BATCH * HEADS), 256, 0, stream>>>(qT, ctxT, hiddenT);

  k_gemm_out<<<dim3(NSP / 128, CIN / 64, BATCH), 256, 0, stream>>>(wo, hiddenT, ybuf);

  k_gn2<<<BATCH * 32, 256, 0, stream>>>(ybuf, gn2_w, gn2_b, out_b, out);
}
